// MultiHeadAttention_27376121545288
// MI455X (gfx1250) — matrix operand order repaired and checked by reading, not compiled, not run on hardware
//
#include <hip/hip_runtime.h>
#include <hip/hip_bf16.h>
#include <math.h>

// ---------------------------------------------------------------------------
// MHA: B=2, S=2048, E=1024, H=16, D=64.  bf16 WMMA pipeline for gfx1250.
// ---------------------------------------------------------------------------

#define BATCH   2
#define SEQ     2048
#define EDIM    1024
#define HEADS   16
#define DHEAD   64
#define MROWS   (BATCH * SEQ)        // 4096

typedef __attribute__((ext_vector_type(8)))  __bf16 v8bf;
typedef __attribute__((ext_vector_type(16))) __bf16 v16bf;
typedef __attribute__((ext_vector_type(8)))  float  v8f;

__device__ __forceinline__ v16bf bf_cat(v8bf lo, v8bf hi) {
    v16bf r;
#pragma unroll
    for (int i = 0; i < 8; ++i) { r[i] = lo[i]; r[8 + i] = hi[i]; }
    return r;
}

__device__ __forceinline__ v8f wmma_bf16(v16bf a, v16bf b, v8f c) {
    return __builtin_amdgcn_wmma_f32_16x16x32_bf16(
        /*neg_a=*/false, a, /*neg_b=*/false, b,
        /*c_mod=*/(short)0, c, /*reuse_a=*/false, /*reuse_b=*/false);
}

// gfx1250 async global->LDS copy of one 16B chunk (ASYNCcnt-tracked).
// Generic pointers to __shared__ carry the LDS byte offset in their low 32 bits.
__device__ __forceinline__ void async_ld16(void* lds_dst, const void* gsrc) {
    unsigned loff = (unsigned)(size_t)lds_dst;
    unsigned long long ga = (unsigned long long)(size_t)gsrc;
    asm volatile("global_load_async_to_lds_b128 %0, %1, off"
                 :: "v"(loff), "v"(ga) : "memory");
}
__device__ __forceinline__ void wait_async0() {
    asm volatile("s_wait_asynccnt 0x0" ::: "memory");
}

// ---------------------------------------------------------------------------
// fp32 -> bf16 pack (n is a multiple of 4)
// ---------------------------------------------------------------------------
__global__ __launch_bounds__(256) void cvt_f32_bf16(
    const float* __restrict__ src, __bf16* __restrict__ dst, int n) {
    int i = (blockIdx.x * 256 + threadIdx.x) * 4;
    if (i < n) {
        float4 f = *(const float4*)(src + i);
        dst[i + 0] = (__bf16)f.x;
        dst[i + 1] = (__bf16)f.y;
        dst[i + 2] = (__bf16)f.z;
        dst[i + 3] = (__bf16)f.w;
    }
}

// ---------------------------------------------------------------------------
// GEMM:  C(M,N) = A(MxK) * W(NxK)^T + bias(N)
//   A, W bf16 row-major (both contract over contiguous K).
//   mode 0: Q  -> bf16 (B,H,S,D), value *= scale (1/sqrt(D)*log2(e))
//   mode 1: K  -> bf16 (B,H,S,D)
//   mode 2: V  -> bf16 (B,H,D,S)  (transposed for the PV wmma B-fragment)
//   mode 3: out-> f32  (M,N) row-major
// Block = 256 threads = 8 waves (4x2); each wave owns a 32x64 tile
// (2 A-frags x 4 B-frags -> 8 wmma per k-step).  Block tile = 128 x 128.
// ---------------------------------------------------------------------------
__global__ __launch_bounds__(256) void gemm_bf16(
    const __bf16* __restrict__ A, const __bf16* __restrict__ W,
    const float* __restrict__ bias, void* __restrict__ Cout,
    int M, int N, int K, int mode, float scale) {
    const int wave = threadIdx.x >> 5;
    const int lane = threadIdx.x & 31;
    const int half = lane >> 4;
    const int r16  = lane & 15;
    const int wm = wave & 3, wn = wave >> 2;
    const int m0 = blockIdx.x * 128 + wm * 32;
    const int n0 = blockIdx.y * 128 + wn * 64;

    v8f acc[2][4] = {};

    for (int k0 = 0; k0 < K; k0 += 32) {
        v16bf af[2], bf[4];
#pragma unroll
        for (int i = 0; i < 2; ++i) {
            // A-fragment (16x32): lane holds row m, K groups {half*8..+7, 16+half*8..+7}
            const __bf16* pa = A + (size_t)(m0 + i * 16 + r16) * K + k0 + half * 8;
            af[i] = bf_cat(*(const v8bf*)pa, *(const v8bf*)(pa + 16));
        }
#pragma unroll
        for (int j = 0; j < 4; ++j) {
            // B-fragment (32x16 of W^T): lane holds W row n, K groups {half*8..+7, 16+half*8..+7}
            const __bf16* pb = W + (size_t)(n0 + j * 16 + r16) * K + k0 + half * 8;
            bf[j] = bf_cat(*(const v8bf*)pb, *(const v8bf*)(pb + 16));
        }
#pragma unroll
        for (int i = 0; i < 2; ++i)
#pragma unroll
            for (int j = 0; j < 4; ++j)
                acc[i][j] = wmma_bf16(af[i], bf[j], acc[i][j]);
    }

    // Epilogue.  C-fragment: VGPR v, lane l -> row = v + 8*(l/16), col = l%16.
#pragma unroll
    for (int i = 0; i < 2; ++i)
#pragma unroll
        for (int j = 0; j < 4; ++j)
#pragma unroll
            for (int v = 0; v < 8; ++v) {
                int m = m0 + i * 16 + v + 8 * half;
                int n = n0 + j * 16 + r16;
                float val = acc[i][j][v] + bias[n];
                if (mode == 3) {
                    ((float*)Cout)[(size_t)m * N + n] = val;
                } else {
                    __bf16 bv = (__bf16)(val * scale);
                    int b = m >> 11, s = m & 2047;   // m = b*2048 + s
                    int h = n >> 6,  d = n & 63;     // n = h*64 + d
                    size_t off;
                    if (mode == 2)
                        off = (((size_t)(b * HEADS + h)) * DHEAD + d) * SEQ + s;   // V^T
                    else
                        off = (((size_t)(b * HEADS + h)) * SEQ + s) * DHEAD + d;   // Q / K
                    ((__bf16*)Cout)[off] = bv;
                }
            }
}

// ---------------------------------------------------------------------------
// Flash attention.  Block = 128 threads (4 waves), one wave per 16-query tile,
// all four waves share one (b,h).  K/V 32-key tiles are double-buffered in LDS
// via global_load_async_to_lds_b128 (ASYNCcnt), fragments come from ds loads.
// Q: (B,H,S,D) bf16 pre-scaled by 1/sqrt(D)*log2e; K: (B,H,S,D); Vt: (B,H,D,S).
// ---------------------------------------------------------------------------
__global__ __launch_bounds__(128) void flash_attn(
    const __bf16* __restrict__ Q, const __bf16* __restrict__ Km,
    const __bf16* __restrict__ Vt, __bf16* __restrict__ Oout) {
    __shared__ __align__(16) __bf16 kbuf[2][32 * DHEAD];   // 2 x 4KB: 32 keys x 64 d
    __shared__ __align__(16) __bf16 vbuf[2][DHEAD * 32];   // 2 x 4KB: 64 d x 32 keys
    __shared__ __align__(16) __bf16 plds[4][16 * 32];      // per-wave P bounce

    const int tid  = threadIdx.x;
    const int wave = tid >> 5;
    const int lane = tid & 31;
    const int half = lane >> 4;
    const int r16  = lane & 15;
    const int bh = blockIdx.y;                 // b*HEADS + h
    const int qt = blockIdx.x * 4 + wave;      // 16-query tile index (0..127)

    const __bf16* Qb = Q  + ((size_t)bh * SEQ + qt * 16) * DHEAD;
    const __bf16* Kb = Km + (size_t)bh * SEQ * DHEAD;
    const __bf16* Vb = Vt + (size_t)bh * DHEAD * SEQ;
    __bf16* pw = &plds[wave][0];

    // Q A-fragments for the two d-halves (d 0..31, 32..63), fixed all loop.
    v16bf qa[2];
#pragma unroll
    for (int g = 0; g < 2; ++g) {
        const __bf16* p = Qb + (size_t)r16 * DHEAD + g * 32 + half * 8;
        qa[g] = bf_cat(*(const v8bf*)p, *(const v8bf*)(p + 16));
    }

    v8f o[4] = {};
    float mrow[8], lrow[8];
#pragma unroll
    for (int v = 0; v < 8; ++v) { mrow[v] = -1e30f; lrow[v] = 0.0f; }

    const int NT = SEQ / 32;                   // 64 key tiles
    const int vrow = tid >> 1;                 // V staging: 2 threads per d-row
    const int vcb  = (tid & 1) * 32;           // byte offset within the 64B row

    // Stage tile 0 into buffer 0.
    {
        const char* ks = (const char*)Kb;                       // tile 0: contiguous 4KB
        char*       kd = (char*)&kbuf[0][0];
        async_ld16(kd + tid * 16,        ks + tid * 16);
        async_ld16(kd + tid * 16 + 2048, ks + tid * 16 + 2048);
        const char* vs = (const char*)(Vb + (size_t)vrow * SEQ) + vcb;
        char*       vd = (char*)&vbuf[0][0] + vrow * 64 + vcb;
        async_ld16(vd,      vs);
        async_ld16(vd + 16, vs + 16);
    }
    wait_async0();
    __syncthreads();

    for (int kt = 0; kt < NT; ++kt) {
        // Prefetch next tile into the other buffer (async, overlapped).
        if (kt + 1 < NT) {
            const int nb = (kt + 1) & 1;
            const char* ks = (const char*)(Kb + (size_t)(kt + 1) * 32 * DHEAD);
            char*       kd = (char*)&kbuf[nb][0];
            async_ld16(kd + tid * 16,        ks + tid * 16);
            async_ld16(kd + tid * 16 + 2048, ks + tid * 16 + 2048);
            const char* vs = (const char*)(Vb + (size_t)vrow * SEQ + (kt + 1) * 32) + vcb;
            char*       vd = (char*)&vbuf[nb][0] + vrow * 64 + vcb;
            async_ld16(vd,      vs);
            async_ld16(vd + 16, vs + 16);
        }

        const __bf16* kb = &kbuf[kt & 1][0];
        const __bf16* vb = &vbuf[kt & 1][0];

        // Scores: queries x 32 keys (two 16-key sub-tiles), frags from LDS.
        v8f s0 = {}, s1 = {};
#pragma unroll
        for (int g = 0; g < 2; ++g) {
            const __bf16* p0 = kb + (size_t)r16 * DHEAD + g * 32 + half * 8;
            v16bf b0 = bf_cat(*(const v8bf*)p0, *(const v8bf*)(p0 + 16));
            const __bf16* p1 = kb + (size_t)(16 + r16) * DHEAD + g * 32 + half * 8;
            v16bf b1 = bf_cat(*(const v8bf*)p1, *(const v8bf*)(p1 + 16));
            s0 = wmma_bf16(qa[g], b0, s0);
            s1 = wmma_bf16(qa[g], b1, s1);
        }

        // Online softmax: row = v + 8*half, N spread over 16 lanes.
        float mt[8];
#pragma unroll
        for (int v = 0; v < 8; ++v) mt[v] = fmaxf(s0[v], s1[v]);
#pragma unroll
        for (int off = 8; off > 0; off >>= 1)
#pragma unroll
            for (int v = 0; v < 8; ++v)
                mt[v] = fmaxf(mt[v], __shfl_xor(mt[v], off, 16));

        float resc[8];
#pragma unroll
        for (int v = 0; v < 8; ++v) {
            float mn = fmaxf(mrow[v], mt[v]);
            resc[v] = exp2f(mrow[v] - mn);      // scale folded into Q already
            mrow[v] = mn;
        }

        float rs[8];
#pragma unroll
        for (int v = 0; v < 8; ++v) {
            float p0 = exp2f(s0[v] - mrow[v]);
            float p1 = exp2f(s1[v] - mrow[v]);
            rs[v] = p0 + p1;
            int row = v + 8 * half;
            pw[row * 32 + r16]      = (__bf16)p0;
            pw[row * 32 + 16 + r16] = (__bf16)p1;
        }
#pragma unroll
        for (int off = 8; off > 0; off >>= 1)
#pragma unroll
            for (int v = 0; v < 8; ++v)
                rs[v] += __shfl_xor(rs[v], off, 16);
#pragma unroll
        for (int v = 0; v < 8; ++v) lrow[v] = lrow[v] * resc[v] + rs[v];

#pragma unroll
        for (int t = 0; t < 4; ++t)
#pragma unroll
            for (int v = 0; v < 8; ++v) o[t][v] *= resc[v];

        // Per-wave LDS RAW (C-layout -> A-layout relayout of P).
        asm volatile("s_wait_dscnt 0x0" ::: "memory");
        const __bf16* pl = pw + r16 * 32 + half * 8;
        v16bf pf = bf_cat(*(const v8bf*)pl, *(const v8bf*)(pl + 16));

        // PV: B-fragment lane holds V^T row d, key groups {half*8..+7, 16+half*8..+7}, from LDS.
#pragma unroll
        for (int t = 0; t < 4; ++t) {
            const __bf16* pv = vb + (size_t)(t * 16 + r16) * 32 + half * 8;
            v16bf vf = bf_cat(*(const v8bf*)pv, *(const v8bf*)(pv + 16));
            o[t] = wmma_bf16(pf, vf, o[t]);
        }

        // Next tile's async copies must have landed; everyone done with current.
        wait_async0();
        __syncthreads();
    }

    // Normalize and store to (B,S,E) bf16 for the out-projection.
    const int b = bh >> 4, h = bh & 15;
#pragma unroll
    for (int v = 0; v < 8; ++v) {
        float inv = 1.0f / lrow[v];
        int s = qt * 16 + v + 8 * half;
#pragma unroll
        for (int t = 0; t < 4; ++t) {
            int e = h * DHEAD + t * 16 + r16;
            Oout[((size_t)b * SEQ + s) * EDIM + e] = (__bf16)(o[t][v] * inv);
        }
    }
}

// ---------------------------------------------------------------------------
// Host-side orchestration
// ---------------------------------------------------------------------------
extern "C" void kernel_launch(void* const* d_in, const int* in_sizes, int n_in,
                              void* d_out, int out_size, void* d_ws, size_t ws_size,
                              hipStream_t stream) {
    (void)in_sizes; (void)n_in; (void)out_size; (void)ws_size;
    const float* q_f  = (const float*)d_in[0];
    const float* k_f  = (const float*)d_in[1];
    const float* v_f  = (const float*)d_in[2];
    const float* win  = (const float*)d_in[3];   // (3E, E)
    const float* bin  = (const float*)d_in[4];   // (3E)
    const float* wout = (const float*)d_in[5];   // (E, E)
    const float* bout = (const float*)d_in[6];   // (E)

    const size_t ACT  = (size_t)MROWS * EDIM;                 // 4M elements
    const size_t WQKV = (size_t)3 * EDIM * EDIM;              // 3M elements
    const size_t WO   = (size_t)EDIM * EDIM;                  // 1M elements
    const size_t BHSD = (size_t)BATCH * HEADS * SEQ * DHEAD;  // 4M elements

    char* ws = (char*)d_ws;
    __bf16* qbf  = (__bf16*)ws;                ws += ACT  * 2;
    __bf16* kbf  = (__bf16*)ws;                ws += ACT  * 2;
    __bf16* vbf  = (__bf16*)ws;                ws += ACT  * 2;
    __bf16* wbf  = (__bf16*)ws;                ws += WQKV * 2;   // Wq|Wk|Wv
    __bf16* wobf = (__bf16*)ws;                ws += WO   * 2;
    __bf16* Qh   = (__bf16*)ws;                ws += BHSD * 2;   // (B,H,S,D)
    __bf16* Kh   = (__bf16*)ws;                ws += BHSD * 2;   // (B,H,S,D)
    __bf16* Vth  = (__bf16*)ws;                ws += BHSD * 2;   // (B,H,D,S)
    __bf16* AO   = (__bf16*)ws;                ws += ACT  * 2;   // (B,S,E)

    // 1) fp32 -> bf16 conversions
    cvt_f32_bf16<<<(int)(ACT  / 1024), 256, 0, stream>>>(q_f,  qbf,  (int)ACT);
    cvt_f32_bf16<<<(int)(ACT  / 1024), 256, 0, stream>>>(k_f,  kbf,  (int)ACT);
    cvt_f32_bf16<<<(int)(ACT  / 1024), 256, 0, stream>>>(v_f,  vbf,  (int)ACT);
    cvt_f32_bf16<<<(int)(WQKV / 1024), 256, 0, stream>>>(win,  wbf,  (int)WQKV);
    cvt_f32_bf16<<<(int)(WO   / 1024), 256, 0, stream>>>(wout, wobf, (int)WO);

    // 2) projections; Q scale folds softmax 1/sqrt(D) and log2(e) for exp2
    dim3 ggrid(MROWS / 128, EDIM / 128);
    gemm_bf16<<<ggrid, 256, 0, stream>>>(qbf, wbf,          bin,            Qh,
                                         MROWS, EDIM, EDIM, 0, 0.18033688011112042f);
    gemm_bf16<<<ggrid, 256, 0, stream>>>(kbf, wbf + WO,     bin + EDIM,     Kh,
                                         MROWS, EDIM, EDIM, 1, 1.0f);
    gemm_bf16<<<ggrid, 256, 0, stream>>>(vbf, wbf + 2 * WO, bin + 2 * EDIM, Vth,
                                         MROWS, EDIM, EDIM, 2, 1.0f);

    // 3) flash attention: grid = (q-tiles/4, B*H)
    flash_attn<<<dim3((SEQ / 16) / 4, BATCH * HEADS), 128, 0, stream>>>(Qh, Kh, Vth, AO);

    // 4) output projection -> fp32 d_out
    gemm_bf16<<<ggrid, 256, 0, stream>>>(AO, wobf, bout, d_out,
                                         MROWS, EDIM, EDIM, 3, 1.0f);
}
